// RoPEMultiheadAttention_83408264888324
// MI455X (gfx1250) — matrix operand order repaired and checked by reading, not compiled, not run on hardware
//
#include <hip/hip_runtime.h>
#include <hip/hip_bf16.h>
#include <math.h>

// Problem constants
#define BB 2
#define TT 2048
#define DD 1024
#define HH 16
#define HD 64

typedef __attribute__((ext_vector_type(16))) __bf16 v16bf;
typedef __attribute__((ext_vector_type(8)))  __bf16 v8bf;
typedef __attribute__((ext_vector_type(4)))  __bf16 v4bf;
typedef __attribute__((ext_vector_type(8)))  float  v8f;
typedef __attribute__((ext_vector_type(4)))  float  v4f;

__device__ __forceinline__ __bf16 f2bf(float f) {
    union { float f; unsigned u; } a; a.f = f;
    unsigned r = a.u + 0x7FFFu + ((a.u >> 16) & 1u);   // round-to-nearest-even
    union { unsigned short s; __bf16 b; } o; o.s = (unsigned short)(r >> 16);
    return o.b;
}

__device__ __forceinline__ v8f wmma_bf16(v16bf a, v16bf b, v8f c) {
    return __builtin_amdgcn_wmma_f32_16x16x32_bf16(false, a, false, b,
                                                   (short)0, c, false, false);
}

// A-fragment (16x32 bf16): lane half h: e[0..15]=row[K=h*16..h*16+15]
__device__ __forceinline__ v16bf frag_a(const __bf16* row, int hlf) {
    union { v16bf v; v8bf h[2]; } u;
    u.h[0] = *(const v8bf*)(row + hlf * 16);
    u.h[1] = *(const v8bf*)(row + hlf * 16 + 8);
    return u.v;
}
// B-fragment: 16 contiguous K values for this lane's column
__device__ __forceinline__ v16bf frag_b16(const __bf16* p) {
    union { v16bf v; v8bf h[2]; } u;
    u.h[0] = *(const v8bf*)(p);
    u.h[1] = *(const v8bf*)(p + 8);
    return u.v;
}

// ---- CDNA5 async global->LDS (ASYNCcnt) --------------------------------
// Per-lane 16B copy straight into LDS, bypassing VGPRs. Low 32 bits of a
// generic LDS pointer are the LDS byte address (ISA aperture rules).
__device__ __forceinline__ void async_ld_b128(const void* g, void* lds) {
    unsigned l = (unsigned)(size_t)lds;
    asm volatile("global_load_async_to_lds_b128 %0, %1, off"
                 :: "v"(l), "v"(g) : "memory");
}
__device__ __forceinline__ void wait_async0() {
#if __has_builtin(__builtin_amdgcn_s_wait_asynccnt)
    __builtin_amdgcn_s_wait_asynccnt(0);
#else
    asm volatile("s_wait_asynccnt 0" ::: "memory");
#endif
}

// ---------------------------------------------------------------- cast fp32 -> bf16
__global__ __launch_bounds__(256)
void cast_bf16(const float* __restrict__ s, __bf16* __restrict__ d, int n4) {
    for (int i = blockIdx.x * 256 + threadIdx.x; i < n4; i += gridDim.x * 256) {
        v4f x = ((const v4f*)s)[i];
        v4bf y;
        for (int j = 0; j < 4; j++) y[j] = f2bf(x[j]);
        ((v4bf*)d)[i] = y;
    }
}

// ---------------------------------------------------------------- WMMA GEMM
// C[M,N] = A[M,K] * B[K,N], bf16 in, f32 accumulate. Double-buffered LDS,
// async tile prefetch overlapped with WMMA compute.
// MODE 0: store f32 C to Cout.
// MODE 1: QKV epilogue: split columns into Q|K|V, apply RoPE to Q,K, scatter
//         bf16 into [B,H,T,hd] buffers.
template<int MODE>
__global__ __launch_bounds__(256)
void gemm_wmma(const __bf16* __restrict__ A, const __bf16* __restrict__ Bm,
               int M, int N, int K,
               float* __restrict__ Cout,
               __bf16* __restrict__ Qo, __bf16* __restrict__ Ko,
               __bf16* __restrict__ Vo)
{
    __shared__ __align__(16) __bf16 As[2][128][32];
    __shared__ __align__(16) __bf16 Bs[2][128][32];   // transposed: Bs[.][n][k]

    const int tid = threadIdx.x, lane = tid & 31, wid = tid >> 5;
    const int hlf = lane >> 4, l15 = lane & 15;
    const int wm = wid & 3, wn = wid >> 2;            // 4 x 2 wave grid
    const int m0 = blockIdx.y * 128, n0 = blockIdx.x * 128;

    auto stage = [&](int k0, int buf) {
        // A tile: per-lane async 16B copies straight into LDS (ASYNCcnt)
        for (int i = 0; i < 2; i++) {
            int c = tid + i * 256;
            int row = c >> 2, kc = (c & 3) * 8;
            async_ld_b128(&A[(size_t)(m0 + row) * K + k0 + kc], &As[buf][row][kc]);
        }
        // B tile transposed: vector global read, 2B LDS scatter
        for (int i = 0; i < 2; i++) {
            int c = tid + i * 256;
            int k = c >> 4, nb = (c & 15) * 8;
            v8bf g = *(const v8bf*)&Bm[(size_t)(k0 + k) * N + n0 + nb];
            for (int j = 0; j < 8; j++) Bs[buf][nb + j][k] = g[j];
        }
    };

    v8f acc[2][4];
    for (int i = 0; i < 2; i++)
        for (int j = 0; j < 4; j++)
            for (int r = 0; r < 8; r++) acc[i][j][r] = 0.f;

    stage(0, 0);
    wait_async0();
    __syncthreads();

    const int NT = K / 32;
    for (int it = 0; it < NT; it++) {
        const int buf = it & 1;
        if (it + 1 < NT) stage((it + 1) * 32, buf ^ 1);   // overlap with WMMA

        v16bf a0 = frag_a(&As[buf][wm * 32 + l15][0], hlf);
        v16bf a1 = frag_a(&As[buf][wm * 32 + 16 + l15][0], hlf);
        for (int nf = 0; nf < 4; nf++) {
            v16bf b = frag_b16(&Bs[buf][wn * 64 + nf * 16 + l15][hlf * 16]);
            acc[0][nf] = wmma_bf16(a0, b, acc[0][nf]);
            acc[1][nf] = wmma_bf16(a1, b, acc[1][nf]);
        }
        wait_async0();
        __syncthreads();
    }

    // epilogue
    for (int fm = 0; fm < 2; fm++)
        for (int nf = 0; nf < 4; nf++)
            for (int r = 0; r < 8; r++) {
                float v = acc[fm][nf][r];
                int m = m0 + wm * 32 + fm * 16 + r + 8 * hlf;
                int c = n0 + wn * 64 + nf * 16 + l15;
                if (MODE == 0) {
                    Cout[(size_t)m * N + c] = v;
                } else {
                    // RoPE partner (feature d^1) lives in adjacent lane
                    float partner = __shfl_xor(v, 1, 32);
                    int b   = m >> 11, t = m & (TT - 1);
                    int sec = c >> 10;                 // 0=Q 1=K 2=V
                    int cc  = c & (DD - 1);
                    int h   = cc >> 6, d = cc & 63;
                    float outv = v;
                    if (sec < 2) {
                        // theta = t * ROPE_BASE^{-(d%32)/32}
                        float theta = (float)t *
                            __expf(-0.28782313662425572f * (float)(d & 31));
                        float sn, cs; __sincosf(theta, &sn, &cs);
                        float rh = (d & 1) ? partner : -partner;
                        outv = v * cs + rh * sn;
                    }
                    __bf16* dst = (sec == 0) ? Qo : (sec == 1) ? Ko : Vo;
                    dst[((((size_t)b * HH + h) * TT) + t) * HD + d] = f2bf(outv);
                }
            }
}

// ---------------------------------------------------------------- flash attention
// One block = one (b,h) head, 64 query rows. 4 waves, 16 query rows per wave.
// Double-buffered K/V tiles, async prefetch overlapped with QK^T/softmax/PV.
__global__ __launch_bounds__(128)
void attn_wmma(const __bf16* __restrict__ Q, const __bf16* __restrict__ K,
               const __bf16* __restrict__ V, __bf16* __restrict__ O)
{
    __shared__ __align__(16) __bf16 Ks [2][64][64];   // [key][hd]
    __shared__ __align__(16) __bf16 Vts[2][64][64];   // [hd][key] (transposed)
    __shared__ __align__(16) __bf16 Ps [4][16][64];   // per-wave P scratch

    const int tid = threadIdx.x, lane = tid & 31, wid = tid >> 5;
    const int hlf = lane >> 4, l15 = lane & 15;
    const int bh = blockIdx.y;
    const int b = bh >> 4, h = bh & 15;
    const int q0 = blockIdx.x * 64;
    const size_t base = (size_t)bh * TT * HD;
    const __bf16* Qp = Q + base;
    const __bf16* Kp = K + base;
    const __bf16* Vp = V + base;

    auto stage = [&](int s0, int buf) {
        // K tile: async per-lane 16B copies into LDS
        for (int i = 0; i < 4; i++) {
            int c = tid + i * 128;
            int row = c >> 3, kc = (c & 7) * 8;
            async_ld_b128(&Kp[(size_t)(s0 + row) * HD + kc], &Ks[buf][row][kc]);
        }
        // V tile transposed: vector global read, 2B LDS scatter
        for (int i = 0; i < 4; i++) {
            int c = tid + i * 128;
            int j = c >> 3, db = (c & 7) * 8;
            v8bf g = *(const v8bf*)&Vp[(size_t)(s0 + j) * HD + db];
            for (int jj = 0; jj < 8; jj++) Vts[buf][db + jj][j] = g[jj];
        }
    };

    // persistent Q fragments for this wave's 16 rows
    const __bf16* qrow = Qp + (size_t)(q0 + wid * 16 + l15) * HD;
    v16bf aq0 = frag_a(qrow, hlf);        // K-window 0..31 of hd
    v16bf aq1 = frag_a(qrow + 32, hlf);   // K-window 32..63

    v8f o[4];
    for (int i = 0; i < 4; i++)
        for (int r = 0; r < 8; r++) o[i][r] = 0.f;
    float mrow[8], lrow[8];
    for (int r = 0; r < 8; r++) { mrow[r] = -1e30f; lrow[r] = 0.f; }

    const float scale = 0.125f;  // 1/sqrt(64)

    stage(0, 0);
    wait_async0();
    __syncthreads();

    const int NT = TT / 64;
    for (int it = 0; it < NT; it++) {
        const int buf = it & 1;
        if (it + 1 < NT) stage((it + 1) * 64, buf ^ 1);   // overlap

        // S = Q * K^T  (B-frag columns = keys, K-dim = hd, contiguous in Ks rows)
        v8f s[4];
        for (int i = 0; i < 4; i++)
            for (int r = 0; r < 8; r++) s[i][r] = 0.f;
        for (int nf = 0; nf < 4; nf++) {
            v16bf kb0 = frag_b16(&Ks[buf][nf * 16 + l15][hlf * 16]);
            v16bf kb1 = frag_b16(&Ks[buf][nf * 16 + l15][32 + hlf * 16]);
            s[nf] = wmma_bf16(aq0, kb0, s[nf]);
            s[nf] = wmma_bf16(aq1, kb1, s[nf]);
        }

        // online softmax (rows live in 16-lane halves -> width-16 reductions)
        float mnew[8], corr[8], rsum[8];
        for (int r = 0; r < 8; r++) {
            float mx = s[0][r] * scale;
            for (int nf = 1; nf < 4; nf++) mx = fmaxf(mx, s[nf][r] * scale);
            mx = fmaxf(mx, __shfl_xor(mx, 1, 16));
            mx = fmaxf(mx, __shfl_xor(mx, 2, 16));
            mx = fmaxf(mx, __shfl_xor(mx, 4, 16));
            mx = fmaxf(mx, __shfl_xor(mx, 8, 16));
            mnew[r] = fmaxf(mrow[r], mx);
            corr[r] = __expf(mrow[r] - mnew[r]);
            rsum[r] = 0.f;
        }
        for (int nf = 0; nf < 4; nf++)
            for (int r = 0; r < 8; r++) {
                float p = __expf(s[nf][r] * scale - mnew[r]);
                rsum[r] += p;
                Ps[wid][r + 8 * hlf][nf * 16 + l15] = f2bf(p); // C-layout -> LDS
            }
        for (int r = 0; r < 8; r++) {
            float sm = rsum[r];
            sm += __shfl_xor(sm, 1, 16);
            sm += __shfl_xor(sm, 2, 16);
            sm += __shfl_xor(sm, 4, 16);
            sm += __shfl_xor(sm, 8, 16);
            lrow[r] = lrow[r] * corr[r] + sm;
            mrow[r] = mnew[r];
        }
        for (int nf = 0; nf < 4; nf++)
            for (int r = 0; r < 8; r++) o[nf][r] *= corr[r];

        // O += P * V.  Ps is wave-private; per-wave LDS ops complete in order,
        // so no barrier needed between the P store above and this reload.
        for (int kst = 0; kst < 2; kst++) {
            v16bf ap = frag_a(&Ps[wid][l15][kst * 32], hlf);
            for (int nf = 0; nf < 4; nf++) {
                v16bf bv = frag_b16(&Vts[buf][nf * 16 + l15][kst * 32 + hlf * 16]);
                o[nf] = wmma_bf16(ap, bv, o[nf]);
            }
        }

        wait_async0();
        __syncthreads();
    }

    // normalize and store to [B,T,H*hd] bf16
    for (int nf = 0; nf < 4; nf++)
        for (int r = 0; r < 8; r++) {
            float val = o[nf][r] / lrow[r];
            int t = q0 + wid * 16 + r + 8 * hlf;
            int d = nf * 16 + l15;
            O[((size_t)b * TT + t) * DD + h * HD + d] = f2bf(val);
        }
}

// ---------------------------------------------------------------- launch
extern "C" void kernel_launch(void* const* d_in, const int* in_sizes, int n_in,
                              void* d_out, int out_size, void* d_ws, size_t ws_size,
                              hipStream_t stream)
{
    const float* x    = (const float*)d_in[0];
    const float* Wqkv = (const float*)d_in[1];
    const float* Wout = (const float*)d_in[2];
    float* out = (float*)d_out;

    const int M    = BB * TT;   // 4096
    const int Nqkv = 3 * DD;    // 3072
    const int Kd   = DD;        // 1024

    __bf16* xb    = (__bf16*)d_ws;
    __bf16* wqkvb = xb    + (size_t)M  * DD;
    __bf16* woutb = wqkvb + (size_t)DD * Nqkv;
    __bf16* Qb    = woutb + (size_t)DD * DD;
    __bf16* Kb    = Qb    + (size_t)M  * DD;
    __bf16* Vb    = Kb    + (size_t)M  * DD;
    __bf16* Ab    = Vb    + (size_t)M  * DD;   // attention output [M][DD]

    cast_bf16<<<1024, 256, 0, stream>>>(x,    xb,    M * DD / 4);
    cast_bf16<<<1024, 256, 0, stream>>>(Wqkv, wqkvb, DD * Nqkv / 4);
    cast_bf16<<<1024, 256, 0, stream>>>(Wout, woutb, DD * DD / 4);

    gemm_wmma<1><<<dim3(Nqkv / 128, M / 128), 256, 0, stream>>>(
        xb, wqkvb, M, Nqkv, Kd, nullptr, Qb, Kb, Vb);

    attn_wmma<<<dim3(TT / 64, BB * HH), 128, 0, stream>>>(Qb, Kb, Vb, Ab);

    gemm_wmma<0><<<dim3(DD / 128, M / 128), 256, 0, stream>>>(
        Ab, woutb, M, DD, Kd, out, nullptr, nullptr, nullptr);
}
